// SelfWindowAttention_60352880443685
// MI455X (gfx1250) — matrix operand order repaired and checked by reading, not compiled, not run on hardware
//
#include <hip/hip_runtime.h>

typedef __attribute__((ext_vector_type(16))) _Float16 v16h;
typedef __attribute__((ext_vector_type(8)))  _Float16 v8h;
typedef __attribute__((ext_vector_type(8)))  float    v8f;

static constexpr int NB   = 4;      // batch
static constexpr int L    = 4096;   // seq len
static constexpr int DIN  = 128;    // input dim
static constexpr int H    = 4;      // heads
static constexpr int DH   = 32;     // head dim
static constexpr int PAD  = 32;     // W//2, W=65
static constexpr int LP   = L + 2*PAD;   // 4160 padded rows
static constexpr int DM   = H*DH;        // 128 model dim
static constexpr int OUTD = 8;

// Build a 16-half WMMA fragment from two 8-half (16B) loads.
static __device__ __forceinline__ v16h frag2(const _Float16* p0, const _Float16* p1) {
  v8h a = *(const v8h*)p0;
  v8h b = *(const v8h*)p1;
  v16h r;
#pragma unroll
  for (int i = 0; i < 8; ++i) { r[i] = a[i]; r[i + 8] = b[i]; }
  return r;
}

static __device__ __forceinline__ v8f wmma_f16(v16h a, v16h b, v8f c) {
  return __builtin_amdgcn_wmma_f32_16x16x32_f16(false, a, false, b, (short)0, c,
                                                false, false);
}

// ---------------------------------------------------------------------------
// Kernel 0: convert W_q, W_k, W_v (f32) -> f16 contiguously into ws.
// ---------------------------------------------------------------------------
__global__ __launch_bounds__(256) void cvt_kernel(
    const float* __restrict__ Wq, const float* __restrict__ Wk,
    const float* __restrict__ Wv, _Float16* __restrict__ dst) {
  int i = blockIdx.x * 256 + threadIdx.x;      // 0 .. 49151
  if (i < 16384)       dst[i] = (_Float16)Wq[i];
  else if (i < 32768)  dst[i] = (_Float16)Wk[i - 16384];
  else if (i < 49152)  dst[i] = (_Float16)Wv[i - 32768];
}

// ---------------------------------------------------------------------------
// Kernel 1: fused QKV projection over the PADDED sequence.
// Block = 256 threads (8 waves); each block does one 16-row tile.
// Wave w computes output channels [16w, 16w+16) for Q, K, V.
// ---------------------------------------------------------------------------
__global__ __launch_bounds__(256) void qkv_kernel(
    const float* __restrict__ x,
    const _Float16* __restrict__ Wq, const _Float16* __restrict__ Wk,
    const _Float16* __restrict__ Wv,
    const float* __restrict__ bq, const float* __restrict__ bk,
    const float* __restrict__ bv,
    _Float16* __restrict__ Qh, _Float16* __restrict__ Kh,
    _Float16* __restrict__ Vh) {
  __shared__ _Float16 XT[16 * DIN];            // 4 KB x-tile (f16)

  const int bIdx = blockIdx.x / (LP / 16);
  const int tile = blockIdx.x % (LP / 16);
  const int p0   = tile * 16;                  // padded row base
  const int tid  = threadIdx.x;

  // Stage x tile into LDS as f16 (zeros in the halo rows).
  {
    const int e0 = tid * 8;
    const int r  = e0 >> 7, c0 = e0 & 127;
    const int l  = p0 + r - PAD;
    const bool ok = (l >= 0) && (l < L);
    const float* src = x + ((size_t)bIdx * L + (ok ? l : 0)) * DIN + c0;
#pragma unroll
    for (int i = 0; i < 8; ++i)
      XT[r * DIN + c0 + i] = ok ? (_Float16)src[i] : (_Float16)0.0f;
  }
  __syncthreads();

  const int w    = tid >> 5;
  const int lane = tid & 31;
  const int n    = lane & 15;                  // also A-matrix row m
  const int hi   = lane >> 4;

  // A fragments: x tile, 4 k-chunks of 32 (shared by all three projections).
  v16h A[4];
#pragma unroll
  for (int kc = 0; kc < 4; ++kc) {
    const _Float16* base = XT + n * DIN + kc * 32;
    A[kc] = frag2(base + 16 * hi, base + 16 * hi + 8);
  }

  const _Float16* Ws[3] = {Wq, Wk, Wv};
  const float*    bs[3] = {bq, bk, bv};
  _Float16*       Os[3] = {Qh, Kh, Vh};

#pragma unroll
  for (int p = 0; p < 3; ++p) {
    v8f acc = {};
#pragma unroll
    for (int kc = 0; kc < 4; ++kc) {
      // B(k, nn) = W[16w + nn][k]; lane reads 16 contiguous k at 16*hi.
      const _Float16* wb = Ws[p] + (size_t)(w * 16 + n) * DIN + kc * 32 + 16 * hi;
      acc = wmma_f16(A[kc], frag2(wb, wb + 8), acc);
    }
    const float bias = bs[p][w * 16 + n];
    _Float16* dst = Os[p] + ((size_t)bIdx * LP + p0) * DM + w * 16 + n;
#pragma unroll
    for (int v = 0; v < 8; ++v)                // D row m = 8*hi + v
      dst[(size_t)(8 * hi + v) * DM] = (_Float16)(acc[v] + bias);
  }
}

// ---------------------------------------------------------------------------
// Kernel 2: fused windowed attention + output projection.
// Block = 128 threads (4 waves); one block per (batch, 16-query tile);
// wave h handles head h. Key union window for 16 queries = 80 padded rows.
// ---------------------------------------------------------------------------
__global__ __launch_bounds__(128) void attn_kernel(
    const _Float16* __restrict__ Qh, const _Float16* __restrict__ Kh,
    const _Float16* __restrict__ Vh, const float* __restrict__ Wo,
    const float* __restrict__ bo, float* __restrict__ out) {
  __shared__ _Float16 VT[DM * 96];        // V transposed: [col][row], 24 KB
  __shared__ _Float16 PB[H * 16 * 96];    // softmax probs per head, 12 KB
  __shared__ float    OB[16 * DM];        // attention output tile, 8 KB

  const int bIdx = blockIdx.x / (L / 16);
  const int tile = blockIdx.x % (L / 16);
  const int l0   = tile * 16;             // query base (also padded key base)
  const int tid  = threadIdx.x;

  // Stage V^T for rows l0..l0+79 (96-row tile, rows 80..95 zero-filled).
  {
    const int c = tid;                    // 0..127
    const _Float16* vsrc = Vh + ((size_t)bIdx * LP + l0) * DM + c;
#pragma unroll 4
    for (int r = 0; r < 96; ++r)
      VT[c * 96 + r] = (r < 80) ? vsrc[(size_t)r * DM] : (_Float16)0.0f;
  }
  __syncthreads();

  const int h    = tid >> 5;
  const int lane = tid & 31;
  const int n    = lane & 15;
  const int hi   = lane >> 4;

  // Q fragment: 16 queries x 32 (K = D_H = 32 -> single WMMA k-chunk).
  const _Float16* qb = Qh + ((size_t)bIdx * LP + l0 + PAD + n) * DM + h * DH;
  const v16h qa = frag2(qb + 16 * hi, qb + 16 * hi + 8);

  // Scores: 5 x 16x16 tiles covering keys j = 0..79 (padded pos l0+j).
  v8f S[5];
#pragma unroll
  for (int t = 0; t < 5; ++t) {
    const _Float16* kb =
        Kh + ((size_t)bIdx * LP + l0 + t * 16 + n) * DM + h * DH + 16 * hi;
    v8f s = {};
    S[t] = wmma_f16(qa, frag2(kb, kb + 8), s);
  }

  const float scale = 0.17677669529663687f;   // 1/sqrt(32)

  // Mask + row softmax. Row m = 8*hi + v spans the 16 lanes of this half.
#pragma unroll
  for (int v = 0; v < 8; ++v) {
    const int m = 8 * hi + v;
    float sv[5];
#pragma unroll
    for (int t = 0; t < 5; ++t) {
      const int j = t * 16 + n;
      const bool ok = (j >= m) && (j <= m + 64) && (l0 + j >= PAD);
      sv[t] = ok ? S[t][v] * scale : -1.0e9f;
    }
    float mx = sv[0];
#pragma unroll
    for (int t = 1; t < 5; ++t) mx = fmaxf(mx, sv[t]);
#pragma unroll
    for (int off = 8; off >= 1; off >>= 1) mx = fmaxf(mx, __shfl_xor(mx, off, 32));
    float sum = 0.0f;
#pragma unroll
    for (int t = 0; t < 5; ++t) { sv[t] = __expf(sv[t] - mx); sum += sv[t]; }
#pragma unroll
    for (int off = 8; off >= 1; off >>= 1) sum += __shfl_xor(sum, off, 32);
    const float inv = 1.0f / sum;
#pragma unroll
    for (int t = 0; t < 5; ++t) S[t][v] = sv[t] * inv;
  }

  // Probs -> LDS (C-layout to A-layout redistribution), pad cols 80..95 = 0.
  _Float16* pb = PB + h * 16 * 96;
#pragma unroll
  for (int v = 0; v < 8; ++v) {
    const int m = 8 * hi + v;
#pragma unroll
    for (int t = 0; t < 5; ++t) pb[m * 96 + t * 16 + n] = (_Float16)S[t][v];
    pb[m * 96 + 80 + n] = (_Float16)0.0f;
  }
  __syncthreads();

  // O(16x32) = P(16x96) @ V(96x32): 3 k-chunks x 2 n-tiles = 6 WMMAs.
  v8f O0 = {}, O1 = {};
#pragma unroll
  for (int u = 0; u < 3; ++u) {
    const _Float16* ab = pb + n * 96 + u * 32;     // A row m = lane&15
    const v16h pa = frag2(ab + 16 * hi, ab + 16 * hi + 8);
    const _Float16* vb0 = VT + (size_t)(h * DH + n) * 96 + u * 32 + 16 * hi;
    O0 = wmma_f16(pa, frag2(vb0, vb0 + 8), O0);
    const _Float16* vb1 = VT + (size_t)(h * DH + 16 + n) * 96 + u * 32 + 16 * hi;
    O1 = wmma_f16(pa, frag2(vb1, vb1 + 8), O1);
  }
#pragma unroll
  for (int v = 0; v < 8; ++v) {
    const int m = 8 * hi + v;
    OB[m * DM + h * DH + n]      = O0[v];
    OB[m * DM + h * DH + 16 + n] = O1[v];
  }
  __syncthreads();

  // Output projection: 16 rows x 8 outputs, one per thread.
  {
    const int m = tid >> 3, o = tid & 7;
    const float* wrow = Wo + o * DM;
    float acc = bo[o];
#pragma unroll 8
    for (int d = 0; d < DM; ++d) acc = fmaf(OB[m * DM + d], wrow[d], acc);
    out[((size_t)bIdx * L + l0 + m) * OUTD + o] = acc;
  }
}

// ---------------------------------------------------------------------------
extern "C" void kernel_launch(void* const* d_in, const int* in_sizes, int n_in,
                              void* d_out, int out_size, void* d_ws, size_t ws_size,
                              hipStream_t stream) {
  (void)in_sizes; (void)n_in; (void)out_size; (void)ws_size;
  const float* x  = (const float*)d_in[0];
  const float* Wq = (const float*)d_in[1];
  const float* bq = (const float*)d_in[2];
  const float* Wk = (const float*)d_in[3];
  const float* bk = (const float*)d_in[4];
  const float* Wv = (const float*)d_in[5];
  const float* bv = (const float*)d_in[6];
  const float* Wo = (const float*)d_in[7];
  const float* bo = (const float*)d_in[8];

  _Float16* ws  = (_Float16*)d_ws;
  _Float16* Wqh = ws;                              // 16384 halves
  _Float16* Wkh = ws + 16384;
  _Float16* Wvh = ws + 32768;
  _Float16* Qh  = ws + 49152;                      // [NB][LP][DM] f16
  _Float16* Kh  = Qh + (size_t)NB * LP * DM;
  _Float16* Vh  = Kh + (size_t)NB * LP * DM;

  cvt_kernel<<<192, 256, 0, stream>>>(Wq, Wk, Wv, ws);
  qkv_kernel<<<NB * (LP / 16), 256, 0, stream>>>(x, Wqh, Wkh, Wvh, bq, bk, bv,
                                                 Qh, Kh, Vh);
  attn_kernel<<<NB * (L / 16), 128, 0, stream>>>(Qh, Kh, Vh, Wo, bo,
                                                 (float*)d_out);
}
